// TRTAttention_74964359184708
// MI455X (gfx1250) — matrix operand order repaired and checked by reading, not compiled, not run on hardware
//
#include <hip/hip_runtime.h>
#include <hip/hip_bf16.h>

// ---------------------------------------------------------------------------
// Problem constants
// ---------------------------------------------------------------------------
#define DIM    768
#define NHEAD  12
#define HD     64
#define HH     48
#define WW     48
#define LTOK   (HH * WW)          // 2304
#define QKVN   (3 * DIM)          // 2304
#define SCALE  0.125f             // 64^-0.5

typedef __attribute__((ext_vector_type(16))) __bf16          v16bf;
typedef __attribute__((ext_vector_type(8)))  float           v8f;
typedef __attribute__((ext_vector_type(8)))  unsigned short  su8;
typedef __attribute__((ext_vector_type(4)))  unsigned int    ui4;
typedef __attribute__((ext_vector_type(8)))  int             i8v;
typedef __attribute__((ext_vector_type(4)))  int             i4v;
typedef unsigned short u16t;

// float -> bf16 bits (round to nearest even)
__device__ __forceinline__ u16t f2bf(float f) {
    unsigned u = __builtin_bit_cast(unsigned, f);
    unsigned r = u + 0x7FFFu + ((u >> 16) & 1u);
    return (u16t)(r >> 16);
}
__device__ __forceinline__ float bf2f(u16t h) {
    unsigned u = ((unsigned)h) << 16;
    return __builtin_bit_cast(float, u);
}

// load a v16bf WMMA operand from two 16-byte LDS chunks
__device__ __forceinline__ v16bf ldop(const u16t* p0, const u16t* p1) {
    union { v16bf v; su8 h[2]; } u;
    u.h[0] = *(const su8*)p0;
    u.h[1] = *(const su8*)p1;
    return u.v;
}

__device__ __forceinline__ v8f wmma_bf16(v16bf a, v16bf b, v8f c) {
    return __builtin_amdgcn_wmma_f32_16x16x32_bf16(
        /*neg_a=*/false, a, /*neg_b=*/false, b,
        /*c_mod=*/(short)0, c, /*reuse_a=*/false, /*reuse_b=*/false);
}

// ---------------------------------------------------------------------------
// Tensor Data Mover: load a 2-D tile of 2-byte elements into LDS.
//   lds_byte_addr : LDS destination byte address
//   gaddr         : global address of tile start
//   tile_w/tile_h : tile dims in elements (tile_w*2 bytes contiguous per row)
//   stride_elems  : row stride in elements
// D# packing per CDNA5 ISA.
// ---------------------------------------------------------------------------
__device__ __forceinline__ void tdm_load_2d_b16(unsigned lds_byte_addr,
                                                const void* gaddr,
                                                unsigned tile_w, unsigned tile_h,
                                                unsigned stride_elems) {
    unsigned long long ga = (unsigned long long)gaddr;
    const unsigned tdim = 0x40000000u;   // huge tensor dims: no OOB clipping

    ui4 g0;
    g0[0] = 1u;                                            // count=1 (valid), user mode
    g0[1] = lds_byte_addr;                                 // lds_addr
    g0[2] = (unsigned)ga;                                  // global_addr[31:0]
    g0[3] = (unsigned)((ga >> 32) & 0x01FFFFFFu)           // global_addr[56:32]
          | (2u << 30);                                    // type=2 ("image")

    i8v g1 = {};
    g1[0] = (int)(1u << 16);                               // data_size=1 -> 2 bytes
    g1[1] = (int)((tdim & 0xFFFFu) << 16);                 // tensor_dim0[15:0]
    g1[2] = (int)(((tdim >> 16) & 0xFFFFu)                 // tensor_dim0[31:16]
          | ((tdim & 0xFFFFu) << 16));                     // tensor_dim1[15:0]
    g1[3] = (int)(((tdim >> 16) & 0xFFFFu)                 // tensor_dim1[31:16]
          | ((tile_w & 0xFFFFu) << 16));                   // tile_dim0
    g1[4] = (int)(tile_h & 0xFFFFu);                       // tile_dim1 (tile_dim2=0)
    g1[5] = (int)stride_elems;                             // tensor_dim0_stride[31:0]
    g1[6] = 0;                                             // stride[47:32], dim1_stride lo
    g1[7] = 0;

    i4v z4 = {};
#if defined(__clang_major__) && (__clang_major__ >= 23)
    i8v z8 = {};
    __builtin_amdgcn_tensor_load_to_lds(g0, g1, z4, z4, z8, 0);
#else
    __builtin_amdgcn_tensor_load_to_lds(g0, g1, z4, z4, 0);
#endif
}

// low 32 bits of a generic shared-aperture address == LDS byte offset
__device__ __forceinline__ unsigned lds_addr_of(const void* p) {
    return (unsigned)(unsigned long long)p;
}

// ---------------------------------------------------------------------------
// Kernel 1/5: generic tiled GEMM, fp32 in/out, bf16 WMMA compute.
// C[M][N] = A[M][K] @ B[K][N] + bias[N].  M,N % 64 == 0, K % 32 == 0.
// Block tile 64x64, 128 threads (4 waves, 2x2 wave grid, 32x32 per wave).
// ---------------------------------------------------------------------------
__global__ __launch_bounds__(128)
void wmma_gemm_bf16_kernel(const float* __restrict__ A,
                           const float* __restrict__ B,
                           const float* __restrict__ bias,
                           float* __restrict__ C,
                           int M, int N, int K) {
    __shared__ u16t lA[64][32];   // [m][k] row-major
    __shared__ u16t lBT[64][32];  // [n][k] (B transposed)

    const int tid  = threadIdx.x;
    const int wave = tid >> 5;
    const int lane = tid & 31;
    const int l16  = lane & 15;
    const int hi   = lane >> 4;

    const int bm = blockIdx.y * 64;
    const int bn = blockIdx.x * 64;
    const int wm = (wave >> 1) * 32;
    const int wn = (wave & 1) * 32;

    v8f acc[2][2] = {};

    for (int k0 = 0; k0 < K; k0 += 32) {
        // stage A tile (64x32), coalesced over k
        for (int i = tid; i < 64 * 32; i += 128) {
            int r = i >> 5, c = i & 31;
            lA[r][c] = f2bf(A[(size_t)(bm + r) * K + k0 + c]);
        }
        // stage B tile transposed (read coalesced over n, write [n][k])
        for (int i = tid; i < 32 * 64; i += 128) {
            int kk = i >> 6, c = i & 63;
            lBT[c][kk] = f2bf(B[(size_t)(k0 + kk) * N + bn + c]);
        }
        __syncthreads();

        v16bf a[2], b[2];
        const int o0 = hi ? 16 : 0, o1 = hi ? 24 : 8;  // A K-chunk layout (contiguous k = 16h + s)
#pragma unroll
        for (int mt = 0; mt < 2; ++mt) {
            const u16t* src = &lA[wm + mt * 16 + l16][0];
            a[mt] = ldop(src + o0, src + o1);
        }
#pragma unroll
        for (int nt = 0; nt < 2; ++nt) {
            const u16t* src = &lBT[wn + nt * 16 + l16][hi * 16];
            b[nt] = ldop(src, src + 8);
        }
#pragma unroll
        for (int mt = 0; mt < 2; ++mt)
#pragma unroll
            for (int nt = 0; nt < 2; ++nt)
                acc[mt][nt] = wmma_bf16(a[mt], b[nt], acc[mt][nt]);
        __syncthreads();
    }

    // epilogue: C/D layout -> row = r + 8*hi within each 16-tile, col = l16
#pragma unroll
    for (int mt = 0; mt < 2; ++mt)
#pragma unroll
        for (int nt = 0; nt < 2; ++nt)
#pragma unroll
            for (int r = 0; r < 8; ++r) {
                int row = bm + wm + mt * 16 + r + 8 * hi;
                int col = bn + wn + nt * 16 + l16;
                C[(size_t)row * N + col] = acc[mt][nt][r] + bias[col];
            }
}

// ---------------------------------------------------------------------------
// Kernel 2/5: split QKV, apply RoPE to q,k, repack to head-major bf16.
// q_bf,k_bf : [h][l][64]   v_t : [h][64][l] (transposed for P@V B-operand)
// ---------------------------------------------------------------------------
__global__ __launch_bounds__(128)
void rope_split_kernel(const float* __restrict__ qkv,
                       u16t* __restrict__ qb, u16t* __restrict__ kb,
                       u16t* __restrict__ vt) {
    int idx = blockIdx.x * blockDim.x + threadIdx.x;   // h*L + l
    if (idx >= NHEAD * LTOK) return;
    int h = idx / LTOK;
    int l = idx - h * LTOK;
    int tx = l % WW;
    int ty = l / WW;

    const float* qp = qkv + (size_t)l * QKVN + h * HD;
    const float* kp = qp + DIM;
    const float* vp = qp + 2 * DIM;
    u16t* qo = qb + (size_t)idx * HD;
    u16t* ko = kb + (size_t)idx * HD;

    const float ln_theta = 9.210340371976184f;  // ln(10000)
#pragma unroll
    for (int j = 0; j < 32; ++j) {
        int fi = j & 15;
        float freq = __expf(-(4.0f * (float)fi / 64.0f) * ln_theta);
        float ang  = (j < 16 ? (float)tx : (float)ty) * freq;
        float c = __cosf(ang), s = __sinf(ang);
        float q0 = qp[2 * j], q1 = qp[2 * j + 1];
        float k0 = kp[2 * j], k1 = kp[2 * j + 1];
        qo[2 * j]     = f2bf(q0 * c - q1 * s);
        qo[2 * j + 1] = f2bf(q0 * s + q1 * c);
        ko[2 * j]     = f2bf(k0 * c - k1 * s);
        ko[2 * j + 1] = f2bf(k0 * s + k1 * c);
    }
#pragma unroll 8
    for (int d = 0; d < HD; ++d)
        vt[((size_t)h * HD + d) * LTOK + l] = f2bf(vp[d]);
}

// ---------------------------------------------------------------------------
// Kernel 3/5: decomposed relative-position bias.
// rel_h[h][l][kh] = sum_c q[h][l][c] * rel_pos_h[qh-kh+47][c]   (and rel_w)
// One block per (h,l); 96 threads: 48 do rel_h, 48 do rel_w.
// ---------------------------------------------------------------------------
__global__ __launch_bounds__(96)
void relbias_kernel(const u16t* __restrict__ qb,
                    const float* __restrict__ rph,
                    const float* __restrict__ rpw,
                    float* __restrict__ relh, float* __restrict__ relw) {
    __shared__ float qrow[HD];
    int hl  = blockIdx.x;            // h*L + l
    int l   = hl % LTOK;
    int tid = threadIdx.x;
    if (tid < HD) qrow[tid] = bf2f(qb[(size_t)hl * HD + tid]);
    __syncthreads();

    int qh = l / WW, qw = l % WW;
    if (tid < HH) {
        const float* rp = rph + (size_t)(qh - tid + HH - 1) * HD;
        float s = 0.f;
#pragma unroll 8
        for (int d = 0; d < HD; ++d) s += qrow[d] * rp[d];
        relh[(size_t)hl * HH + tid] = s;
    } else {
        int kw = tid - HH;
        const float* rp = rpw + (size_t)(qw - kw + WW - 1) * HD;
        float s = 0.f;
#pragma unroll 8
        for (int d = 0; d < HD; ++d) s += qrow[d] * rp[d];
        relw[(size_t)hl * WW + kw] = s;
    }
}

// ---------------------------------------------------------------------------
// Kernel 4/5: flash attention. grid = (L/64, NHEAD), 128 threads (4 waves).
// Q/K/V tiles staged by the Tensor Data Mover (wave 0 issues, TENSORcnt
// waited, workgroup barrier publishes). Each wave owns 16 q-rows. Per 64-key
// block: 8 WMMAs for S=QK^T, online softmax (shfl reductions inside 16-lane
// halves, matching the C/D VGPR row layout), 8 WMMAs for O+=PV.
// ---------------------------------------------------------------------------
__global__ __launch_bounds__(128)
void flash_attn_kernel(const u16t* __restrict__ qb,
                       const u16t* __restrict__ kb,
                       const u16t* __restrict__ vt,
                       const float* __restrict__ relh,
                       const float* __restrict__ relw,
                       float* __restrict__ outp /* [L][768] */) {
    __shared__ u16t lQ[64][64];       // [qrow][d]
    __shared__ u16t lK[64][64];       // [krow][d]
    __shared__ u16t lV[64][64];       // [d][kcol]  (V^T tile)
    __shared__ u16t lP[4][16][64];    // per-wave P tile

    const int h    = blockIdx.y;
    const int qb0  = blockIdx.x * 64;
    const int tid  = threadIdx.x;
    const int wave = tid >> 5;
    const int lane = tid & 31;
    const int l16  = lane & 15;
    const int hi   = lane >> 4;
    const int o0   = hi ? 16 : 0, o1 = hi ? 24 : 8;  // A-operand K chunks (contiguous k = 16h + s)

    // TDM: Q tile (fully contiguous 8KB -> 1-D copy); completion folded into
    // the first k-block's tensorcnt wait.
    if (wave == 0) {
        tdm_load_2d_b16(lds_addr_of(&lQ[0][0]),
                        qb + ((size_t)h * LTOK + qb0) * HD,
                        4096u, 1u, 4096u);
    }

    v8f O[4] = {};
    float mrow[8], srow[8];
#pragma unroll
    for (int r = 0; r < 8; ++r) { mrow[r] = -3.0e38f; srow[r] = 0.f; }

    const int qrow0 = qb0 + wave * 16;
    const size_t relbase = (size_t)h * LTOK;

    for (int kbk = 0; kbk < LTOK / 64; ++kbk) {
        const int k0 = kbk * 64;
        __syncthreads();   // all waves done reading lK/lV of previous block
        if (wave == 0) {
            // K tile: 64 rows x 64 dims, contiguous -> 1-D 8KB copy
            tdm_load_2d_b16(lds_addr_of(&lK[0][0]),
                            kb + ((size_t)h * LTOK + k0) * HD,
                            4096u, 1u, 4096u);
            // V^T tile: 64 hd-rows x 64 key-cols, row stride L
            tdm_load_2d_b16(lds_addr_of(&lV[0][0]),
                            vt + (size_t)h * HD * LTOK + k0,
                            64u, 64u, (unsigned)LTOK);
            __builtin_amdgcn_s_wait_tensorcnt(0);
        }
        __syncthreads();

        // ---- S = Q @ K^T  (16 x 64 per wave) ----
        v8f S[4] = {};
#pragma unroll
        for (int ks = 0; ks < 64; ks += 32) {
            const u16t* asrc = &lQ[wave * 16 + l16][ks];
            v16bf aop = ldop(asrc + o0, asrc + o1);
#pragma unroll
            for (int nt = 0; nt < 4; ++nt) {
                const u16t* bsrc = &lK[nt * 16 + l16][ks + hi * 16];
                S[nt] = wmma_bf16(aop, ldop(bsrc, bsrc + 8), S[nt]);
            }
        }

        // ---- scale + rel bias + online softmax ----
        float p[4][8];
#pragma unroll
        for (int r = 0; r < 8; ++r) {
            const int row = qrow0 + r + 8 * hi;
            float rm = -3.0e38f;
#pragma unroll
            for (int nt = 0; nt < 4; ++nt) {
                int col = k0 + nt * 16 + l16;
                float v = S[nt][r] * SCALE
                        + relh[(relbase + row) * HH + col / WW]
                        + relw[(relbase + row) * WW + col % WW];
                p[nt][r] = v;
                rm = fmaxf(rm, v);
            }
            // reduce row-max across the 16 lanes of this half-wave
            for (int m = 1; m < 16; m <<= 1) rm = fmaxf(rm, __shfl_xor(rm, m, 32));
            float mo = mrow[r];
            float mn = fmaxf(mo, rm);
            float corr = __expf(mo - mn);
            float rs = 0.f;
#pragma unroll
            for (int nt = 0; nt < 4; ++nt) {
                float e = __expf(p[nt][r] - mn);
                p[nt][r] = e;
                rs += e;
            }
            for (int m = 1; m < 16; m <<= 1) rs += __shfl_xor(rs, m, 32);
            srow[r] = srow[r] * corr + rs;
            mrow[r] = mn;
#pragma unroll
            for (int nt = 0; nt < 4; ++nt) O[nt][r] *= corr;
            // spill P to LDS (re-layout C/D form -> A-operand form)
#pragma unroll
            for (int nt = 0; nt < 4; ++nt)
                lP[wave][r + 8 * hi][nt * 16 + l16] = f2bf(p[nt][r]);
        }
        asm volatile("s_wait_dscnt 0" ::: "memory");  // own-wave LDS RAW

        // ---- O += P @ V  (16 x 64 per wave, K = 64 keys) ----
#pragma unroll
        for (int ks = 0; ks < 64; ks += 32) {
            const u16t* asrc = &lP[wave][l16][ks];
            v16bf aop = ldop(asrc + o0, asrc + o1);
#pragma unroll
            for (int nt = 0; nt < 4; ++nt) {
                const u16t* bsrc = &lV[nt * 16 + l16][ks + hi * 16];
                O[nt] = wmma_bf16(aop, ldop(bsrc, bsrc + 8), O[nt]);
            }
        }
    }

    // ---- epilogue: normalize, write attn output [L][nh*hd] ----
#pragma unroll
    for (int nt = 0; nt < 4; ++nt)
#pragma unroll
        for (int r = 0; r < 8; ++r) {
            int row = qrow0 + r + 8 * hi;
            int col = nt * 16 + l16;
            outp[(size_t)row * DIM + h * HD + col] = O[nt][r] / srow[r];
        }
}

// ---------------------------------------------------------------------------
// Host-side launch
// ---------------------------------------------------------------------------
extern "C" void kernel_launch(void* const* d_in, const int* in_sizes, int n_in,
                              void* d_out, int out_size, void* d_ws, size_t ws_size,
                              hipStream_t stream) {
    (void)in_sizes; (void)n_in; (void)out_size; (void)ws_size;
    const float* x      = (const float*)d_in[0];  // [L][768]
    const float* w_qkv  = (const float*)d_in[1];  // [768][2304]
    const float* b_qkv  = (const float*)d_in[2];  // [2304]
    const float* w_proj = (const float*)d_in[3];  // [768][768]
    const float* b_proj = (const float*)d_in[4];  // [768]
    const float* rph    = (const float*)d_in[5];  // [95][64]
    const float* rpw    = (const float*)d_in[6];  // [95][64]
    float* out = (float*)d_out;                   // [L][768]

    // workspace carve-up (256B aligned)
    char* ws = (char*)d_ws;
    size_t off = 0;
    auto carve = [&](size_t bytes) {
        void* p = ws + off;
        off += (bytes + 255) & ~(size_t)255;
        return p;
    };
    float* qkv     = (float*)carve((size_t)LTOK * QKVN * 4);           // 21.2 MB
    u16t*  q_bf    = (u16t*) carve((size_t)NHEAD * LTOK * HD * 2);     // 3.5 MB
    u16t*  k_bf    = (u16t*) carve((size_t)NHEAD * LTOK * HD * 2);     // 3.5 MB
    u16t*  v_t     = (u16t*) carve((size_t)NHEAD * LTOK * HD * 2);     // 3.5 MB
    float* relh    = (float*)carve((size_t)NHEAD * LTOK * HH * 4);     // 5.3 MB
    float* relw    = (float*)carve((size_t)NHEAD * LTOK * WW * 4);     // 5.3 MB
    float* attn_o  = (float*)carve((size_t)LTOK * DIM * 4);            // 7.1 MB

    // 1) QKV = X @ W_qkv + b
    wmma_gemm_bf16_kernel<<<dim3(QKVN / 64, LTOK / 64), 128, 0, stream>>>(
        x, w_qkv, b_qkv, qkv, LTOK, QKVN, DIM);

    // 2) RoPE + head-major repack (q,k bf16; v transposed bf16)
    {
        int n = NHEAD * LTOK;
        rope_split_kernel<<<(n + 127) / 128, 128, 0, stream>>>(qkv, q_bf, k_bf, v_t);
    }

    // 3) decomposed relative-position bias tables
    relbias_kernel<<<NHEAD * LTOK, 96, 0, stream>>>(q_bf, rph, rpw, relh, relw);

    // 4) flash attention -> attn_o [L][768]
    flash_attn_kernel<<<dim3(LTOK / 64, NHEAD), 128, 0, stream>>>(
        q_bf, k_bf, v_t, relh, relw, attn_o);

    // 5) out = attn_o @ W_proj + b_proj
    wmma_gemm_bf16_kernel<<<dim3(DIM / 64, LTOK / 64), 128, 0, stream>>>(
        attn_o, w_proj, b_proj, out, LTOK, DIM, DIM);
}
